// RWKV_RNN_47811575939186
// MI455X (gfx1250) — matrix operand order repaired and checked by reading, not compiled, not run on hardware
//
#include <hip/hip_runtime.h>
#include <hip/hip_bf16.h>

#define NN 1024
#define HH 4096
#define LLAYERS 24
#define VOCAB 50277

typedef __attribute__((ext_vector_type(16))) __bf16 v16bf;
typedef __attribute__((ext_vector_type(8)))  float  v8f;

struct RwkvP {
  const float *emb, *ln0w, *ln0b, *ln1w, *ln1b, *ln2w, *ln2b;
  const float *tmk, *tmv, *tmr, *tf, *td;
  const float *kw, *vw, *rw, *ow;
  const float *ftk, *ftr, *fkw, *fvw, *frw;
  const float *lnow, *lnob, *headw, *state;
  const int   *token;
  float *out;        // [0,V) logits, [V, V+5*L*N) new state
  float *ws;         // x | k | v | rp | rab | fk(H) | rp2
  unsigned *cnt;     // grid barrier counter (zeroed by init kernel)
  int nblk;
};

// Native bf16 convert (gfx1250 has bf16 VALU; clang lowers fptrunc directly).
__device__ __forceinline__ __bf16 f2bf(float f) { return static_cast<__bf16>(f); }

// Device-wide barrier: monotonically increasing counter, agent scope.
__device__ __forceinline__ void gsync(unsigned* cnt, int nblk, unsigned& phase) {
  __syncthreads();
  ++phase;
  if (threadIdx.x == 0) {
    __hip_atomic_fetch_add(cnt, 1u, __ATOMIC_ACQ_REL, __HIP_MEMORY_SCOPE_AGENT);
    const unsigned target = phase * (unsigned)nblk;
    while (__hip_atomic_load(cnt, __ATOMIC_ACQUIRE, __HIP_MEMORY_SCOPE_AGENT) < target)
      __builtin_amdgcn_s_sleep(1);
  }
  __syncthreads();
}

__device__ __forceinline__ float block_sum(float v, float* red) {
  const int tid = threadIdx.x;
  #pragma unroll
  for (int o = 16; o; o >>= 1) v += __shfl_down(v, o, 32);
  if ((tid & 31) == 0) red[tid >> 5] = v;
  __syncthreads();
  if (tid < 32) {                       // warp 0 folds the 8 partials
    v = (tid < 8) ? red[tid] : 0.f;
    #pragma unroll
    for (int o = 4; o; o >>= 1) v += __shfl_down(v, o, 32);
    if (tid == 0) red[0] = v;
  }
  __syncthreads();
  float r = red[0];
  __syncthreads();
  return r;
}

// LayerNorm of 1024-elem global vector into LDS (block-redundant).
__device__ __forceinline__ void ln_to_lds(const float* xg, const float* w,
                                          const float* b, float* dst, float* red) {
  float s = 0.f;
  for (int i = threadIdx.x; i < NN; i += blockDim.x) s += xg[i];
  const float mean = block_sum(s, red) * (1.0f / NN);
  float vs = 0.f;
  for (int i = threadIdx.x; i < NN; i += blockDim.x) { float d = xg[i] - mean; vs += d * d; }
  const float var = block_sum(vs, red) * (1.0f / NN);
  const float rstd = rsqrtf(var + 1e-5f);
  for (int i = threadIdx.x; i < NN; i += blockDim.x)
    dst[i] = (xg[i] - mean) * rstd * w[i] + b[i];
  __syncthreads();
}

// One wave computes dot(W[row,:], v) with b128 coalesced loads; result on lane 0.
__device__ __forceinline__ float dot_row(const float* __restrict__ wrow,
                                         const float* __restrict__ v, int n) {
  const int lane = threadIdx.x & 31;
  float acc0 = 0.f, acc1 = 0.f;
  for (int k = lane * 4; k < n; k += 128) {
    const float4 wv = *reinterpret_cast<const float4*>(wrow + k);
    const float4 xv = *reinterpret_cast<const float4*>(v + k);
    acc0 = fmaf(wv.x, xv.x, acc0); acc1 = fmaf(wv.y, xv.y, acc1);
    acc0 = fmaf(wv.z, xv.z, acc0); acc1 = fmaf(wv.w, xv.w, acc1);
  }
  float acc = acc0 + acc1;
  #pragma unroll
  for (int o = 16; o; o >>= 1) acc += __shfl_down(acc, o, 32);
  return acc;
}

__global__ void __launch_bounds__(256)
rwkv_fused_kernel(RwkvP p) {
  __shared__ __align__(16) float sm[4 * NN];   // xx|xk|xv|xr, or fk staging (4096)
  __shared__ float red[32];
  __shared__ __align__(32) __bf16 xbf[NN];

  float* smA = sm;            // xx / xx2 / head-LN
  float* smB = sm + NN;       // xk / fxk
  float* smC = sm + 2 * NN;   // xv / fxr
  float* smD = sm + 3 * NN;   // xr

  float* ws_x   = p.ws;
  float* ws_k   = p.ws + NN;
  float* ws_v   = p.ws + 2 * NN;
  float* ws_rp  = p.ws + 3 * NN;
  float* ws_rab = p.ws + 4 * NN;
  float* ws_fk  = p.ws + 5 * NN;          // H floats
  float* ws_rp2 = p.ws + 5 * NN + HH;

  unsigned phase = 0;
  const int tid  = threadIdx.x;
  const int lane = tid & 31;
  const int wid  = tid >> 5;
  const int gw   = blockIdx.x * (blockDim.x >> 5) + wid;   // global wave id
  const int nW   = gridDim.x * (blockDim.x >> 5);
  const int gth  = blockIdx.x * blockDim.x + tid;
  const int nT   = gridDim.x * blockDim.x;

  // ---- init: x = LN0(emb[token]) ----
  if (blockIdx.x == 0) {
    const float* e = p.emb + (size_t)p.token[0] * NN;
    ln_to_lds(e, p.ln0w, p.ln0b, smA, red);
    for (int i = tid; i < NN; i += blockDim.x) ws_x[i] = smA[i];
  }
  gsync(p.cnt, p.nblk, phase);

  for (int l = 0; l < LLAYERS; ++l) {
    const float* st    = p.state + (size_t)l * 5 * NN;
    float*       stOut = p.out + VOCAB + (size_t)l * 5 * NN;

    // ---- time-mix prologue: LN1 + mixing into LDS (block-redundant) ----
    ln_to_lds(ws_x, p.ln1w + l * NN, p.ln1b + l * NN, smA, red);
    for (int i = tid; i < NN; i += blockDim.x) {
      const float xx = smA[i];
      const float sa = st[NN + i];
      const float tk = p.tmk[l * NN + i], tv = p.tmv[l * NN + i], tr = p.tmr[l * NN + i];
      smB[i] = xx * tk + sa * (1.f - tk);
      smC[i] = xx * tv + sa * (1.f - tv);
      smD[i] = xx * tr + sa * (1.f - tr);
      if (blockIdx.x == 0) stOut[NN + i] = xx;          // new_s[1] = xx
    }
    __syncthreads();

    // ---- k/v/r GEMVs: 3N rows over all waves ----
    const float* kwl = p.kw + (size_t)l * NN * NN;
    const float* vwl = p.vw + (size_t)l * NN * NN;
    const float* rwl = p.rw + (size_t)l * NN * NN;
    for (int row = gw; row < 3 * NN; row += nW) {
      const int m = row >> 10, r = row & (NN - 1);
      const float* W = (m == 0 ? kwl : (m == 1 ? vwl : rwl)) + (size_t)r * NN;
      const float* xv = (m == 0 ? smB : (m == 1 ? smC : smD));
      const float d = dot_row(W, xv, NN);
      if (lane == 0) (m == 0 ? ws_k : (m == 1 ? ws_v : ws_rp))[r] = d;
    }
    gsync(p.cnt, p.nblk, phase);

    // ---- WKV recurrence (elementwise, one owner per element) ----
    for (int i = gth; i < NN; i += nT) {
      const float k = ws_k[i], v = ws_v[i], rp = ws_rp[i];
      const float aa = st[2 * NN + i], bb = st[3 * NN + i], pp = st[4 * NN + i];
      const float ww = p.tf[l * NN + i] + k;
      const float q  = fmaxf(pp, ww);
      const float e1 = __expf(pp - q), e2 = __expf(ww - q);
      const float a  = e1 * aa + e2 * v;
      const float b  = e1 * bb + e2;
      const float r  = 1.f / (1.f + __expf(-rp));
      ws_rab[i] = r * a / b;
      const float ww2 = pp + p.td[l * NN + i];
      const float q2  = fmaxf(ww2, k);
      const float e1n = __expf(ww2 - q2), e2n = __expf(k - q2);
      stOut[2 * NN + i] = e1n * aa + e2n * v;           // aa_n
      stOut[3 * NN + i] = e1n * bb + e2n;               // bb_n
      stOut[4 * NN + i] = q2;                           // pp_n
    }
    gsync(p.cnt, p.nblk, phase);

    // ---- x += ow @ (r*a/b) ----
    for (int i = tid; i < NN; i += blockDim.x) smA[i] = ws_rab[i];
    __syncthreads();
    const float* owl = p.ow + (size_t)l * NN * NN;
    for (int row = gw; row < NN; row += nW) {
      const float d = dot_row(owl + (size_t)row * NN, smA, NN);
      if (lane == 0) ws_x[row] += d;
    }
    gsync(p.cnt, p.nblk, phase);

    // ---- channel-mix prologue: LN2 + mixing ----
    ln_to_lds(ws_x, p.ln2w + l * NN, p.ln2b + l * NN, smA, red);
    for (int i = tid; i < NN; i += blockDim.x) {
      const float xx2 = smA[i];
      const float sf  = st[i];
      const float tk = p.ftk[l * NN + i], tr = p.ftr[l * NN + i];
      smB[i] = xx2 * tk + sf * (1.f - tk);
      smC[i] = xx2 * tr + sf * (1.f - tr);
      if (blockIdx.x == 0) stOut[i] = xx2;              // new_s[0] = xx2
    }
    __syncthreads();

    // ---- fk = relu(fkw@fxk)^2 (H rows) and rp2 = frw@fxr (N rows) ----
    const float* fkwl = p.fkw + (size_t)l * HH * NN;
    const float* frwl = p.frw + (size_t)l * NN * NN;
    for (int row = gw; row < HH + NN; row += nW) {
      if (row < HH) {
        float d = dot_row(fkwl + (size_t)row * NN, smB, NN);
        d = fmaxf(d, 0.f);
        if (lane == 0) ws_fk[row] = d * d;
      } else {
        const int r = row - HH;
        const float d = dot_row(frwl + (size_t)r * NN, smC, NN);
        if (lane == 0) ws_rp2[r] = d;
      }
    }
    gsync(p.cnt, p.nblk, phase);

    // ---- x += sigmoid(rp2) * (fvw @ fk), fk staged to LDS ----
    for (int i = tid; i < HH; i += blockDim.x) sm[i] = ws_fk[i];
    __syncthreads();
    const float* fvwl = p.fvw + (size_t)l * NN * HH;
    for (int row = gw; row < NN; row += nW) {
      const float d = dot_row(fvwl + (size_t)row * HH, sm, HH);
      if (lane == 0) {
        const float r = 1.f / (1.f + __expf(-ws_rp2[row]));
        ws_x[row] += r * d;
      }
    }
    gsync(p.cnt, p.nblk, phase);
  }

  // ---- head: logits = head_w @ LN(x)  via v_wmma_f32_16x16x32_bf16 ----
  ln_to_lds(ws_x, p.lnow, p.lnob, smA, red);
  for (int i = tid; i < NN; i += blockDim.x) xbf[i] = f2bf(smA[i]);
  __syncthreads();

  const int ngroups = (VOCAB + 15) >> 4;
  const int koff = (lane >> 4) << 4;   // 0 | 16 : A fragment K offset for lane half
  const int boff = (lane >> 4) << 4;   // 0 | 16 : B fragment K offset for lane half
  for (int g = gw; g < ngroups; g += nW) {
    const int r0 = g << 4;
    int row = r0 + (lane & 15);
    if (row >= VOCAB) row = VOCAB - 1;               // clamp; tail rows never written
    const float* wr = p.headw + (size_t)row * NN;
    v8f c = {0.f, 0.f, 0.f, 0.f, 0.f, 0.f, 0.f, 0.f};
    #pragma unroll 2
    for (int k0 = 0; k0 < NN; k0 += 32) {
      // A: 16x32 bf16 tile of head_w. Lane m + lane m+16 cover 128 contiguous
      // bytes of row m -> fully line-coalesced b128 loads.
      const float4* w0 = reinterpret_cast<const float4*>(wr + k0 + koff);
      const float4* w1 = reinterpret_cast<const float4*>(wr + k0 + koff + 8);
      const float4 a0 = w0[0], a1 = w0[1], a2 = w1[0], a3 = w1[1];
      v16bf a;
      a[0]  = f2bf(a0.x); a[1]  = f2bf(a0.y); a[2]  = f2bf(a0.z); a[3]  = f2bf(a0.w);
      a[4]  = f2bf(a1.x); a[5]  = f2bf(a1.y); a[6]  = f2bf(a1.z); a[7]  = f2bf(a1.w);
      a[8]  = f2bf(a2.x); a[9]  = f2bf(a2.y); a[10] = f2bf(a2.z); a[11] = f2bf(a2.w);
      a[12] = f2bf(a3.x); a[13] = f2bf(a3.y); a[14] = f2bf(a3.z); a[15] = f2bf(a3.w);
      // B: x chunk broadcast to all 16 columns (lanes<16: K=k0..k0+15, lanes>=16: +16)
      v16bf b;
      __builtin_memcpy(&b, &xbf[k0 + boff], sizeof(b));
      c = __builtin_amdgcn_wmma_f32_16x16x32_bf16(false, a, false, b,
                                                  (short)0, c, false, false);
    }
    // D layout: VGPR j, lane 0 -> M=j ; lane 16 -> M=8+j (broadcast B => any column)
    if ((lane & 15) == 0) {                           // lanes 0 and 16 write 8 rows each
      const int half = lane >> 4;
      float* dst = p.out + r0 + 8 * half;
      if (r0 + 16 <= VOCAB) {                         // fast path: two b128 stores
        const float4 o0 = make_float4(c[0], c[1], c[2], c[3]);
        const float4 o1 = make_float4(c[4], c[5], c[6], c[7]);
        *reinterpret_cast<float4*>(dst)     = o0;
        *reinterpret_cast<float4*>(dst + 4) = o1;
      } else {                                        // tail group only
        #pragma unroll
        for (int j = 0; j < 8; ++j) {
          const int rr = r0 + 8 * half + j;
          if (rr < VOCAB) p.out[rr] = c[j];
        }
      }
    }
  }
}

__global__ void rwkv_cnt_init(unsigned* cnt) { *cnt = 0u; }

extern "C" void kernel_launch(void* const* d_in, const int* in_sizes, int n_in,
                              void* d_out, int out_size, void* d_ws, size_t ws_size,
                              hipStream_t stream) {
  (void)in_sizes; (void)n_in; (void)out_size; (void)ws_size;
  RwkvP p;
  p.emb   = (const float*)d_in[0];
  p.ln0w  = (const float*)d_in[1];  p.ln0b = (const float*)d_in[2];
  p.ln1w  = (const float*)d_in[3];  p.ln1b = (const float*)d_in[4];
  p.ln2w  = (const float*)d_in[5];  p.ln2b = (const float*)d_in[6];
  p.tmk   = (const float*)d_in[7];  p.tmv  = (const float*)d_in[8];
  p.tmr   = (const float*)d_in[9];  p.tf   = (const float*)d_in[10];
  p.td    = (const float*)d_in[11];
  p.kw    = (const float*)d_in[12]; p.vw   = (const float*)d_in[13];
  p.rw    = (const float*)d_in[14]; p.ow   = (const float*)d_in[15];
  p.ftk   = (const float*)d_in[16]; p.ftr  = (const float*)d_in[17];
  p.fkw   = (const float*)d_in[18]; p.fvw  = (const float*)d_in[19];
  p.frw   = (const float*)d_in[20];
  p.lnow  = (const float*)d_in[21]; p.lnob = (const float*)d_in[22];
  p.headw = (const float*)d_in[23];
  p.state = (const float*)d_in[24];
  p.token = (const int*)d_in[25];
  p.out = (float*)d_out;
  p.ws  = (float*)d_ws;
  // float scratch occupies exactly (5N + H + N)*4 = 40960 bytes; counter after it
  p.cnt = (unsigned*)((char*)d_ws + 40960);

  // Persistent grid: exactly what can be co-resident (barrier safety).
  int dev = 0; (void)hipGetDevice(&dev);
  int mp = 0;
  (void)hipDeviceGetAttribute(&mp, hipDeviceAttributeMultiprocessorCount, dev);
  int perMP = 0;
  (void)hipOccupancyMaxActiveBlocksPerMultiprocessor(&perMP, rwkv_fused_kernel, 256, 0);
  if (mp <= 0) mp = 64;
  if (perMP <= 0) perMP = 1;
  long nblk = (long)mp * (long)perMP;
  if (nblk > 2048) nblk = 2048;
  p.nblk = (int)nblk;

  hipLaunchKernelGGL(rwkv_cnt_init, dim3(1), dim3(1), 0, stream, p.cnt);
  hipLaunchKernelGGL(rwkv_fused_kernel, dim3((unsigned)nblk), dim3(256), 0, stream, p);
}
